// ExportableWaveRNN_25331717112510
// MI455X (gfx1250) — matrix operand order repaired and checked by reading, not compiled, not run on hardware
//
#include <hip/hip_runtime.h>
#include <hip/hip_bf16.h>
#include <math.h>

typedef __attribute__((ext_vector_type(16))) __bf16 v16bf;
typedef __attribute__((ext_vector_type(8)))  float  v8f;

#define B_ 16
#define T_ 2000
#define C_ 128
#define G_ 512
#define H_ 512
#define V_ 256

// ---------- fragment loaders (wave32, 16x16x32 bf16 WMMA layouts, ISA 7.12.2) ----------

// A-matrix 16x32 bf16: lane 0-15 = row m, elems 0-7 -> K kb+0..7, elems 8-15 -> K kb+16..23
//                      lane16-31 = row m-16, elems 0-7 -> K kb+8..15, elems 8-15 -> K kb+24..31
__device__ __forceinline__ v16bf load_a_frag(const __bf16* base, int ld, int kb, int lane) {
  const int m  = lane & 15;
  const int kh = (lane >> 4) << 3;          // 0 or 8
  const __bf16* p = base + m * ld + kb + kh;
  union { v16bf v; uint4 q[2]; } r;
  r.q[0] = *(const uint4*)(p);
  r.q[1] = *(const uint4*)(p + 16);
  return r.v;
}

// B-matrix 32x16 bf16 from row-major W[ldb] (row = output neuron j, col = k):
// lane 0-15: col n, K kb+0..7,16..23 ; lane 16-31: col n, K kb+8..15,24..31
__device__ __forceinline__ v16bf load_b_frag(const __bf16* W, int ldb, int j0, int kb, int lane) {
  const int n    = lane & 15;
  const int koff = kb + ((lane >> 4) << 3); // +0 or +8
  const __bf16* p = W + (size_t)(j0 + n) * ldb + koff;
  union { v16bf v; uint4 q[2]; } r;
  r.q[0] = *(const uint4*)(p);
  r.q[1] = *(const uint4*)(p + 16);
  return r.v;
}

#define WMMA_BF16(A, B, Cacc) \
  __builtin_amdgcn_wmma_f32_16x16x32_bf16(false, (A), false, (B), (short)0, (Cacc), false, false)

// ---------- JAX threefry2x32 (key(1) = {0,1}) ----------
__device__ __forceinline__ void tf2x32(unsigned k0, unsigned k1, unsigned x0, unsigned x1,
                                       unsigned& o0, unsigned& o1) {
  const unsigned k2 = k0 ^ k1 ^ 0x1BD11BDAu;
  x0 += k0; x1 += k1;
  auto R = [&](int r) { x0 += x1; x1 = (x1 << r) | (x1 >> (32 - r)); x1 ^= x0; };
  R(13); R(15); R(26); R(6);   x0 += k1; x1 += k2 + 1u;
  R(17); R(29); R(16); R(24);  x0 += k2; x1 += k0 + 2u;
  R(13); R(15); R(26); R(6);   x0 += k0; x1 += k1 + 3u;
  R(17); R(29); R(16); R(24);  x0 += k1; x1 += k2 + 4u;
  R(13); R(15); R(26); R(6);   x0 += k2; x1 += k0 + 5u;
  o0 = x0; o1 = x1;
}

__device__ __forceinline__ float sigmoidf_(float x) { return 1.0f / (1.0f + expf(-x)); }

// ---------- fp32 -> bf16 weight conversion pre-pass ----------
__global__ void cvt_bf16_kernel(const float* __restrict__ src, __bf16* __restrict__ dst, int n) {
  int i = blockIdx.x * blockDim.x + threadIdx.x;
  if (i < n) dst[i] = (__bf16)src[i];
}

// ---------- persistent WaveRNN kernel: 1 block x 1024 threads (32 wave32 waves) ----------
__global__ __launch_bounds__(1024) void wavernn_persist(
    const float* __restrict__ cond, const float* __restrict__ h0,
    const float* __restrict__ b_ih, const float* __restrict__ b_hh,
    const float* __restrict__ b_hid, const float* __restrict__ b_out,
    const float* __restrict__ emb,
    const __bf16* __restrict__ Wih, const __bf16* __restrict__ Whh,
    const __bf16* __restrict__ Whid, const __bf16* __restrict__ Wout,
    float* __restrict__ out) {
  __shared__ __bf16 x_s[B_][C_];     //  4 KB : x_t = cond_t + embedding[prev]
  __shared__ __bf16 h_s[B_][G_];     // 16 KB : GRU hidden state
  __shared__ __bf16 hid_s[B_][H_];   // 16 KB : relu hidden
  __shared__ float  logit_s[B_][V_]; // 16 KB : logits for sampling
  __shared__ int    prev_s[B_];

  const int tid  = threadIdx.x;
  const int wave = tid >> 5;
  const int lane = tid & 31;
  const int n    = lane & 15;          // fragment column
  const int mh   = (lane >> 4) << 3;   // fragment row base (0 or 8)

  // ---- loop-invariant per-lane scalars: load ONCE, live in registers across t ----
  const int g0 = wave << 4;            // gate-phase column tile base (G-space)
  const int g  = g0 + n;
  const float br  = b_ih[g]          + b_hh[g];           // r-gate bias (i+h combined)
  const float bz  = b_ih[G_ + g]     + b_hh[G_ + g];      // z-gate bias (i+h combined)
  const float bin = b_ih[2 * G_ + g];                     // n-gate input bias
  const float bhn = b_hh[2 * G_ + g];                     // n-gate hidden bias
  const int n0 = wave << 4;                               // hid/out tile base
  const float bbh = b_hid[n0 + n];
  const float bbo = b_out[((wave & (V_ / 16 - 1)) << 4) + n];

  // init: h from h0, prev = mid-bucket, x for t=0
  for (int i = tid; i < B_ * G_; i += 1024) (&h_s[0][0])[i] = (__bf16)h0[i];
  if (tid < B_) prev_s[tid] = V_ / 2;
  __syncthreads();
  for (int i = tid; i < B_ * C_; i += 1024) {
    int m = i >> 7, c = i & 127;
    x_s[m][c] = (__bf16)(cond[((size_t)m * T_) * C_ + c] + emb[prev_s[m] * C_ + c]);
  }
  __syncthreads();

  const v8f vzero = {0.f, 0.f, 0.f, 0.f, 0.f, 0.f, 0.f, 0.f};

  for (int t = 0; t < T_; ++t) {
    // Stop LICM from hoisting the (loop-invariant) weight fragment loads out of
    // the 2000-step loop and spilling them to scratch: weights must be
    // re-streamed from the L2-resident bf16 arrays every step.
    asm volatile("" ::: "memory");

    // ======== GRU gates: wave owns columns [g0, g0+16) of G ========
    // r and z gates accumulate i-part and h-part into ONE accumulator (K=640);
    // only the n gate needs i_n and h_n separate (4 live v8f accumulators).
    v8f accr = vzero, accz = vzero, accin = vzero, acchn = vzero;
    #pragma unroll
    for (int kb = 0; kb < C_; kb += 32) {  // x @ W_ih^T
      v16bf a = load_a_frag(&x_s[0][0], C_, kb, lane);
      accr  = WMMA_BF16(a, load_b_frag(Wih, C_, 0 * G_ + g0, kb, lane), accr);
      accz  = WMMA_BF16(a, load_b_frag(Wih, C_, 1 * G_ + g0, kb, lane), accz);
      accin = WMMA_BF16(a, load_b_frag(Wih, C_, 2 * G_ + g0, kb, lane), accin);
    }
    #pragma unroll
    for (int kb = 0; kb < G_; kb += 32) {  // h @ W_hh^T
      v16bf a = load_a_frag(&h_s[0][0], G_, kb, lane);
      accr  = WMMA_BF16(a, load_b_frag(Whh, G_, 0 * G_ + g0, kb, lane), accr);
      accz  = WMMA_BF16(a, load_b_frag(Whh, G_, 1 * G_ + g0, kb, lane), accz);
      acchn = WMMA_BF16(a, load_b_frag(Whh, G_, 2 * G_ + g0, kb, lane), acchn);
    }
    float hnew[8];
    #pragma unroll
    for (int e = 0; e < 8; ++e) {          // fragment elem e -> (m = mh+e, col g)
      float r  = sigmoidf_(accr[e] + br);
      float z  = sigmoidf_(accz[e] + bz);
      float nn = tanhf((accin[e] + bin) + r * (acchn[e] + bhn));
      float ho = (float)h_s[mh + e][g];
      hnew[e]  = (1.0f - z) * nn + z * ho;
    }
    __syncthreads();                       // all waves done reading old h
    #pragma unroll
    for (int e = 0; e < 8; ++e) h_s[mh + e][g] = (__bf16)hnew[e];
    __syncthreads();

    // ======== hid = relu(h @ W_hid^T + b_hid) ========
    {
      v8f acc = vzero;
      #pragma unroll
      for (int kb = 0; kb < G_; kb += 32)
        acc = WMMA_BF16(load_a_frag(&h_s[0][0], G_, kb, lane),
                        load_b_frag(Whid, G_, n0, kb, lane), acc);
      #pragma unroll
      for (int e = 0; e < 8; ++e) {
        float v = acc[e] + bbh;
        hid_s[mh + e][n0 + n] = (__bf16)(v > 0.0f ? v : 0.0f);
      }
    }
    __syncthreads();

    // ======== logits = hid @ W_out^T + b_out (waves 0..15) ========
    if (wave < V_ / 16) {
      v8f acc = vzero;
      #pragma unroll
      for (int kb = 0; kb < H_; kb += 32)
        acc = WMMA_BF16(load_a_frag(&hid_s[0][0], H_, kb, lane),
                        load_b_frag(Wout, H_, n0, kb, lane), acc);
      #pragma unroll
      for (int e = 0; e < 8; ++e) {
        float v = acc[e] + bbo;
        int m = mh + e;
        logit_s[m][n0 + n] = v;
        out[((size_t)m * T_ + t) * V_ + n0 + n] = v;  // logits [B,T,V]
      }
    }
    __syncthreads();

    // ======== Gumbel-max categorical sample (wave m handles batch row m) ========
    if (wave < B_) {
      const int m = wave;
      // child key t of split(key(1), T)  (threefry count layout)
      unsigned kt0, kt1, a0, a1, c0, c1;
      if (t < T_ / 2) {
        tf2x32(0u, 1u, (unsigned)(2 * t),     (unsigned)(T_ + 2 * t),     a0, a1);
        tf2x32(0u, 1u, (unsigned)(2 * t + 1), (unsigned)(T_ + 2 * t + 1), c0, c1);
        kt0 = a0; kt1 = c0;
      } else {
        tf2x32(0u, 1u, (unsigned)(2 * t - T_),     (unsigned)(2 * t),     a0, a1);
        tf2x32(0u, 1u, (unsigned)(2 * t + 1 - T_), (unsigned)(2 * t + 1), c0, c1);
        kt0 = a1; kt1 = c1;
      }
      float bv = -3.4e38f; int bi = 0;
      #pragma unroll
      for (int j = 0; j < V_ / 32; ++j) {
        const int v = lane + 32 * j;
        const unsigned idx  = (unsigned)(m * V_ + v);
        const unsigned half = (unsigned)(B_ * V_ / 2);
        unsigned o0, o1, bits;
        if (idx < half) { tf2x32(kt0, kt1, idx, half + idx, o0, o1); bits = o0; }
        else            { tf2x32(kt0, kt1, idx - half, idx, o0, o1); bits = o1; }
        float u = __uint_as_float(0x3f800000u | (bits >> 9)) - 1.0f;  // [0,1)
        const float tiny = 1.17549435e-38f;
        u = u * (1.0f - tiny) + tiny;
        float val = logit_s[m][v] + (-logf(-logf(u)));
        if (val > bv) { bv = val; bi = v; }
      }
      #pragma unroll
      for (int off = 16; off; off >>= 1) {
        float ov = __shfl_xor(bv, off, 32);
        int   oi = __shfl_xor(bi, off, 32);
        if (ov > bv || (ov == bv && oi < bi)) { bv = ov; bi = oi; }
      }
      if (lane == 0) {
        prev_s[m] = bi;
        out[(size_t)B_ * T_ * V_ + (size_t)m * T_ + t] = (float)bi;  // samples [B,T]
      }
    }
    __syncthreads();

    // ======== x for t+1 = cond + embedding[sample] ========
    if (t + 1 < T_) {
      for (int i = tid; i < B_ * C_; i += 1024) {
        int m = i >> 7, c = i & 127;
        x_s[m][c] = (__bf16)(cond[((size_t)m * T_ + (t + 1)) * C_ + c] +
                             emb[prev_s[m] * C_ + c]);
      }
    }
    __syncthreads();
  }

  // h_final [B,G]
  const size_t hf = (size_t)B_ * T_ * V_ + (size_t)B_ * T_;
  for (int i = tid; i < B_ * G_; i += 1024) out[hf + i] = (float)(&h_s[0][0])[i];
}

// ---------- launcher ----------
extern "C" void kernel_launch(void* const* d_in, const int* in_sizes, int n_in,
                              void* d_out, int out_size, void* d_ws, size_t ws_size,
                              hipStream_t stream) {
  (void)in_sizes; (void)n_in; (void)out_size; (void)ws_size;
  const float* cond  = (const float*)d_in[0];
  const float* h0    = (const float*)d_in[1];
  const float* W_ih  = (const float*)d_in[2];
  const float* W_hh  = (const float*)d_in[3];
  const float* b_ih  = (const float*)d_in[4];
  const float* b_hh  = (const float*)d_in[5];
  const float* W_hid = (const float*)d_in[6];
  const float* b_hid = (const float*)d_in[7];
  const float* W_out = (const float*)d_in[8];
  const float* b_out = (const float*)d_in[9];
  const float* emb   = (const float*)d_in[10];
  float* out = (float*)d_out;

  // bf16 weight cache in workspace (L2-resident working set ~2.62 MB)
  __bf16* ws      = (__bf16*)d_ws;
  __bf16* Wih_bf  = ws;                               // 1536*128
  __bf16* Whh_bf  = Wih_bf + 3 * G_ * C_;             // 1536*512
  __bf16* Whid_bf = Whh_bf + 3 * G_ * G_;             // 512*512
  __bf16* Wout_bf = Whid_bf + H_ * G_;                // 256*512

  const int n_ih  = 3 * G_ * C_;
  const int n_hh  = 3 * G_ * G_;
  const int n_hid = H_ * G_;
  const int n_out = V_ * H_;
  cvt_bf16_kernel<<<(n_ih  + 255) / 256, 256, 0, stream>>>(W_ih,  Wih_bf,  n_ih);
  cvt_bf16_kernel<<<(n_hh  + 255) / 256, 256, 0, stream>>>(W_hh,  Whh_bf,  n_hh);
  cvt_bf16_kernel<<<(n_hid + 255) / 256, 256, 0, stream>>>(W_hid, Whid_bf, n_hid);
  cvt_bf16_kernel<<<(n_out + 255) / 256, 256, 0, stream>>>(W_out, Wout_bf, n_out);

  wavernn_persist<<<1, 1024, 0, stream>>>(cond, h0, b_ih, b_hh, b_hid, b_out, emb,
                                          Wih_bf, Whh_bf, Whid_bf, Wout_bf, out);
}
